// InteractionLayer_32134945309413
// MI455X (gfx1250) — matrix operand order repaired and checked by reading, not compiled, not run on hardware
//
#include <hip/hip_runtime.h>

// N=8192 atoms, D=256 features, cutoff=1, mu=1, sigma=0.5
#define NN 8192
#define DD 256
#define LDS_STRIDE 96   // 32 bf16 = 64B data + 32B pad; keeps 32B alignment for v16bf loads

typedef __attribute__((ext_vector_type(16))) __bf16        v16bf;
typedef __attribute__((ext_vector_type(8)))  float         v8f;
typedef __attribute__((ext_vector_type(4)))  float         f4;
typedef __attribute__((ext_vector_type(4)))  unsigned int  u4;

// ---------------------------------------------------------------------------
// Kernel 1: build swizzled Wt so each msg_kernel A-operand is ONE contiguous
// 32B load.  Row d holds 256 bf16: [c=0..7][lh=0..1][e=0..15] where
//   k = c*32 + lh*8 + e + (e>=8 ? 8 : 0)     (16-bit A-operand lane layout)
// ---------------------------------------------------------------------------
__global__ void wt_kernel(const float* __restrict__ W, __bf16* __restrict__ Wt) {
  int d  = blockIdx.x;
  int o  = threadIdx.x;
  int c  = o >> 5;
  int lh = (o >> 4) & 1;
  int e  = o & 15;
  int k  = c * 32 + lh * 8 + e + ((e >> 3) << 3);
  Wt[d * DD + o] = (__bf16)W[k * DD + d];
}

// ---------------------------------------------------------------------------
// Kernel 2: msgT[d][j] = bf16( sum_k W[k][d]*z[j][k] + bias[d] )
// One wave per 16-wide j tile, all 16 d-tiles (128 accum VGPRs).
// ---------------------------------------------------------------------------
__global__ void __launch_bounds__(256)
msg_kernel(const float* __restrict__ z, const __bf16* __restrict__ Wt,
           const float* __restrict__ bias, __bf16* __restrict__ msgT) {
  const int wave  = (blockIdx.x * blockDim.x + threadIdx.x) >> 5; // 0..511
  const int lane  = threadIdx.x & 31;
  const int ln    = lane & 15;
  const int lh    = lane >> 4;
  const int jBase = wave * 16;

  v8f acc[16] = {};
  const float* zrow = z + (size_t)(jBase + ln) * DD;

  for (int ks = 0; ks < DD; ks += 32) {
    const int c = ks >> 5;
    // B operand: z[j][ks + lh*8 .. +8) then z[j][ks + 16 + lh*8 .. +8)
    const f4* zp = (const f4*)(zrow + ks + lh * 8);
    f4 z0 = zp[0], z1 = zp[1], z2 = zp[4], z3 = zp[5];

    // A operands: 16 tiles, each one 32B contiguous load from swizzled Wt
    v16bf am[16];
#pragma unroll
    for (int t = 0; t < 16; ++t)
      am[t] = *(const v16bf*)(Wt + ((size_t)(t * 16 + ln) * 8 + c) * 32 + lh * 16);

    v16bf bm;
    bm[0]  = (__bf16)z0.x; bm[1]  = (__bf16)z0.y; bm[2]  = (__bf16)z0.z; bm[3]  = (__bf16)z0.w;
    bm[4]  = (__bf16)z1.x; bm[5]  = (__bf16)z1.y; bm[6]  = (__bf16)z1.z; bm[7]  = (__bf16)z1.w;
    bm[8]  = (__bf16)z2.x; bm[9]  = (__bf16)z2.y; bm[10] = (__bf16)z2.z; bm[11] = (__bf16)z2.w;
    bm[12] = (__bf16)z3.x; bm[13] = (__bf16)z3.y; bm[14] = (__bf16)z3.z; bm[15] = (__bf16)z3.w;

#pragma unroll
    for (int t = 0; t < 16; ++t)
      acc[t] = __builtin_amdgcn_wmma_f32_16x16x32_bf16(
          false, am[t], false, bm, (short)0, acc[t], false, false);
  }

  // Epilogue: bias add, store transposed bf16. C/D layout: m = r + 8*lh, n = ln.
#pragma unroll
  for (int t = 0; t < 16; ++t) {
#pragma unroll
    for (int r = 0; r < 8; ++r) {
      int d = t * 16 + r + lh * 8;
      msgT[(size_t)d * NN + jBase + ln] = (__bf16)(acc[t][r] + bias[d]);
    }
  }
}

// ---------------------------------------------------------------------------
// Kernel 3: out[i][d] = sum_j weight(dist[i][j]) * msg[j][d]
// weight(r) = (r < 1) ? exp(-2*(1/r - 1)^2) : 0     (mu=1, 2*sigma^2=0.5)
// Wave = 16 rows x all 256 cols (exp computed ONCE per dist element).
// msg K-chunks double-buffered in LDS via GLOBAL_LOAD_ASYNC_TO_LDS_B128
// (ASYNCcnt), one barrier per iteration.
// ---------------------------------------------------------------------------
__device__ __forceinline__ void async_copy_chunk(
    const __bf16* __restrict__ msgT, unsigned char* buf,
    int srow, int scol, int jcol) {
#pragma unroll
  for (int s = 0; s < 8; ++s) {
    const __bf16* g = msgT + (size_t)(srow + s * 32) * NN + jcol + scol * 8;
    unsigned lds = (unsigned)(unsigned long long)(buf + (srow + s * 32) * LDS_STRIDE + scol * 16);
    // memory -> LDS, no VGPR round trip; tracked by ASYNCcnt
    asm volatile("global_load_async_to_lds_b128 %0, %1, off"
                 :: "v"(lds), "v"(g) : "memory");
  }
}

__global__ void __launch_bounds__(128)
main_kernel(const float* __restrict__ dist, const __bf16* __restrict__ msgT,
            float* __restrict__ out) {
  __shared__ unsigned char smem[2 * 256 * LDS_STRIDE];  // 48 KB, double buffered

  const int tid     = threadIdx.x;
  const int wave    = tid >> 5;
  const int lane    = tid & 31;
  const int ln      = lane & 15;
  const int lh      = lane >> 4;
  const int rowBase = blockIdx.x * 64 + wave * 16;
  const float* drow = dist + (size_t)(rowBase + ln) * NN;

  const int srow = tid >> 2;  // staging rows: srow, srow+32, ..., srow+224
  const int scol = tid & 3;   // 16B column within the 64B row chunk

  v8f acc[16] = {};

  // Kick off async copy of chunk 0 into buffer 0
  async_copy_chunk(msgT, smem, srow, scol, 0);

  for (int jc = 0; jc < NN; jc += 32) {
    unsigned char* buf  = smem + ((jc >> 5) & 1) * (256 * LDS_STRIDE);
    unsigned char* bufN = smem + (((jc >> 5) + 1) & 1) * (256 * LDS_STRIDE);

    // 1) Streaming dist loads for this chunk (LOADcnt — independent of the
    //    ASYNCcnt staging, so the weights' wait is not entangled with it).
    const f4* dp  = (const f4*)(drow + jc + lh * 16);
    const f4* dp2 = (const f4*)(drow + jc + lh * 16 + 8);
    f4 r0 = __builtin_nontemporal_load(dp);
    f4 r1 = __builtin_nontemporal_load(dp + 1);
    f4 r2 = __builtin_nontemporal_load(dp2);
    f4 r3 = __builtin_nontemporal_load(dp2 + 1);

    // 2) This wave's async copies of the current chunk have landed in LDS...
    asm volatile("s_wait_asynccnt 0x0" ::: "memory");
    // ...and after the barrier, every wave's have. The barrier also proves all
    // waves finished reading bufN last iteration, so it is free to overwrite.
    __syncthreads();

    // 3) Kick off async copy of the NEXT chunk into the other buffer
    //    (latency hidden behind the exp/WMMA work below).
    int jn = jc + 32;
    if (jn < NN) async_copy_chunk(msgT, bufN, srow, scol, jn);

    // 4) Issue ALL 16 B-operand tile loads (32B contiguous, 32B aligned)
    v16bf bm[16];
#pragma unroll
    for (int t = 0; t < 16; ++t)
      bm[t] = *(const v16bf*)(buf + (t * 16 + ln) * LDS_STRIDE + lh * 32);

    // 5) Weights on the trans pipe (overlaps DS latency), native bf16 cvt.
    //    A-operand lane layout: halves 0-15 <- K = lh*16.. (contiguous).
    float rv[16] = {r0.x, r0.y, r0.z, r0.w, r1.x, r1.y, r1.z, r1.w,
                    r2.x, r2.y, r2.z, r2.w, r3.x, r3.y, r3.z, r3.w};
    v16bf am;
#pragma unroll
    for (int e = 0; e < 16; ++e) {
      float r   = rv[e];
      float dlt = __builtin_amdgcn_rcpf(r) - 1.0f;                  // v_rcp_f32
      float sv  = __builtin_amdgcn_exp2f(-2.88539008f * dlt * dlt); // v_exp_f32
      am[e] = (__bf16)((r < 1.0f) ? sv : 0.0f);
    }

    // 6) 16 back-to-back WMMAs
#pragma unroll
    for (int t = 0; t < 16; ++t)
      acc[t] = __builtin_amdgcn_wmma_f32_16x16x32_bf16(
          false, am, false, bm[t], (short)0, acc[t], false, false);
  }

  // Store: C/D layout m = r + 8*lh, n = ln; nontemporal (write-once stream)
#pragma unroll
  for (int t = 0; t < 16; ++t) {
#pragma unroll
    for (int r = 0; r < 8; ++r)
      __builtin_nontemporal_store(
          acc[t][r], &out[(size_t)(rowBase + r + lh * 8) * DD + t * 16 + ln]);
  }
}

// ---------------------------------------------------------------------------
extern "C" void kernel_launch(void* const* d_in, const int* in_sizes, int n_in,
                              void* d_out, int out_size, void* d_ws, size_t ws_size,
                              hipStream_t stream) {
  const float* z    = (const float*)d_in[0];  // [8192, 256]
  const float* dist = (const float*)d_in[1];  // [8192, 8192]
  const float* W    = (const float*)d_in[2];  // [256, 256]
  const float* bias = (const float*)d_in[3];  // [256]
  float* out = (float*)d_out;                 // [8192, 256]

  // Workspace: msgT bf16 [256][8192] (4 MB) then swizzled Wt bf16 (128 KB)
  __bf16* msgT = (__bf16*)d_ws;
  __bf16* Wt   = (__bf16*)((char*)d_ws + (size_t)DD * NN * sizeof(__bf16));

  wt_kernel<<<dim3(DD), dim3(DD), 0, stream>>>(W, Wt);
  msg_kernel<<<dim3(64), dim3(256), 0, stream>>>(z, Wt, bias, msgT);      // 512 waves
  main_kernel<<<dim3(NN / 64), dim3(128), 0, stream>>>(dist, msgT, out);  // 128 WGs
}
